// GraphableDecodeStep_16990890623414
// MI455X (gfx1250) — matrix operand order repaired and checked by reading, not compiled, not run on hardware
//
#include <hip/hip_runtime.h>
#include <hip/hip_bf16.h>
#include <math.h>

// ---------------- model constants ----------------
#define L_LAYERS 4
#define HID      2048
#define NH       32
#define NKV      8
#define HD       64
#define INTER    8192
#define VOCAB    32000
#define MAXSEQ   4096

typedef __attribute__((ext_vector_type(16))) _Float16 v16h;
typedef __attribute__((ext_vector_type(8)))  float    v8f;

// ---------------- workspace layout (floats) ----------------
// h:2048 | x:2048 | q:2048 | kn:512 | vn:512 | scores:32*4096 | o:2048 | act:8192 | partials: 32*32000
#define OFF_H    0
#define OFF_X    2048
#define OFF_Q    4096
#define OFF_KN   6144
#define OFF_VN   6656
#define OFF_SC   7168
#define OFF_O    (OFF_SC + NH * MAXSEQ)        // 138240
#define OFF_ACT  (OFF_O + 2048)                // 140288
#define OFF_PART (OFF_ACT + INTER)             // 148480 ; region holds up to 32*32000 floats (~4.1 MB)

// ---------------- kernels ----------------

__global__ void embed_kernel(const int* __restrict__ ids,
                             const float* __restrict__ embed,
                             float* __restrict__ h) {
    int i = blockIdx.x * 256 + threadIdx.x;
    if (i < HID) h[i] = embed[(size_t)ids[0] * HID + i];
}

__global__ void rmsnorm_kernel(const float* __restrict__ in,
                               const float* __restrict__ w,
                               float* __restrict__ out, int n) {
    __shared__ float red[256];
    int tid = threadIdx.x;
    float s = 0.f;
    for (int i = tid; i < n; i += 256) { float v = in[i]; s += v * v; }
    red[tid] = s; __syncthreads();
    for (int st = 128; st > 0; st >>= 1) {
        if (tid < st) red[tid] += red[tid + st];
        __syncthreads();
    }
    float inv = rsqrtf(red[0] / (float)n + 1e-6f);
    for (int i = tid; i < n; i += 256) out[i] = in[i] * inv * w[i];
}

// Split-K GEMV partial. W row-major [K,N]. Each thread produces 4 adjacent outputs
// via float4 (global_load_b128) -> one wave streams 512 contiguous bytes per k-row.
// This is the HBM-bound core of the workload (~1.3 GB of weights per step).
__global__ void gemv_partial_kernel(const float* __restrict__ x,
                                    const float* __restrict__ W,
                                    float* __restrict__ part,
                                    int K, int N, int kChunk) {
    __shared__ float xs[256];
    int j4 = (blockIdx.x * 256 + threadIdx.x) * 4;
    int k0 = blockIdx.y * kChunk;
    for (int i = threadIdx.x; i < kChunk; i += 256) xs[i] = x[k0 + i];
    __syncthreads();
    if (j4 >= N) return;
    float4 acc = make_float4(0.f, 0.f, 0.f, 0.f);
    const float* Wp = W + (size_t)k0 * N + j4;
    #pragma unroll 4
    for (int i = 0; i < kChunk; ++i) {
        float4 w = *(const float4*)(Wp + (size_t)i * N);
        float xv = xs[i];
        acc.x = fmaf(xv, w.x, acc.x);
        acc.y = fmaf(xv, w.y, acc.y);
        acc.z = fmaf(xv, w.z, acc.z);
        acc.w = fmaf(xv, w.w, acc.w);
    }
    *(float4*)(part + (size_t)blockIdx.y * N + j4) = acc;
}

__global__ void reduce_kernel(const float* __restrict__ part,
                              float* __restrict__ out,
                              int N, int nCh, int accumulate) {
    int j4 = (blockIdx.x * 256 + threadIdx.x) * 4;
    if (j4 >= N) return;
    float4 s = make_float4(0.f, 0.f, 0.f, 0.f);
    for (int c = 0; c < nCh; ++c) {
        float4 p = *(const float4*)(part + (size_t)c * N + j4);
        s.x += p.x; s.y += p.y; s.z += p.z; s.w += p.w;
    }
    float4* op = (float4*)(out + j4);
    if (accumulate) {
        float4 ov = *op;
        s.x += ov.x; s.y += ov.y; s.z += ov.z; s.w += ov.w;
    }
    *op = s;
}

__global__ void silu_mul_reduce_kernel(const float* __restrict__ pg,
                                       const float* __restrict__ pu,
                                       float* __restrict__ act,
                                       int N, int nCh) {
    int j4 = (blockIdx.x * 256 + threadIdx.x) * 4;
    if (j4 >= N) return;
    float4 gs = make_float4(0.f, 0.f, 0.f, 0.f);
    float4 us = make_float4(0.f, 0.f, 0.f, 0.f);
    for (int c = 0; c < nCh; ++c) {
        float4 pgv = *(const float4*)(pg + (size_t)c * N + j4);
        float4 puv = *(const float4*)(pu + (size_t)c * N + j4);
        gs.x += pgv.x; gs.y += pgv.y; gs.z += pgv.z; gs.w += pgv.w;
        us.x += puv.x; us.y += puv.y; us.z += puv.z; us.w += puv.w;
    }
    float4 r;
    r.x = gs.x / (1.0f + expf(-gs.x)) * us.x;
    r.y = gs.y / (1.0f + expf(-gs.y)) * us.y;
    r.z = gs.z / (1.0f + expf(-gs.z)) * us.z;
    r.w = gs.w / (1.0f + expf(-gs.w)) * us.w;
    *(float4*)(act + j4) = r;
}

// RoPE on q (32 heads) and new k (8 heads), in place. 1280 pairs.
__global__ void rope_kernel(const int* __restrict__ pos,
                            float* __restrict__ q,
                            float* __restrict__ kn) {
    int idx = blockIdx.x * 256 + threadIdx.x;
    if (idx >= (NH + NKV) * (HD / 2)) return;
    int head = idx / (HD / 2);
    int d    = idx % (HD / 2);
    float p = (float)pos[0];
    float inv_freq = powf(10000.0f, -(float)d / (float)(HD / 2));
    float f = p * inv_freq;
    float c = cosf(f), s = sinf(f);
    float* base = (head < NH) ? (q + head * HD) : (kn + (head - NH) * HD);
    float a = base[d], b = base[d + HD / 2];
    base[d]          = a * c - b * s;
    base[d + HD / 2] = b * c + a * s;
}

// Attention scores via V_WMMA_F32_16X16X32_F16.
// Per KV head g: C(16x16) = Q(16x64,f16) x K^T(64x16,f16), rows 0..3 valid (4 q-heads / group).
// One wave per 16-timestep tile; grid = (32 blocks of 8 waves, NKV).
// A layout (ISA 7.12.2, 16-bit A 16x32): lane = row M = lane&15, half = lane>>4;
//   element e -> k = (e<8 ? half*8+e : 16 + half*8 + (e-8)).
// B layout (32x16): lane = column N = lane&15, half = lane>>4; element e -> k same as A.
// C layout: lanes 0..15 VGPR v -> (M=v, N=lane).
__global__ void attn_scores_kernel(const float* __restrict__ q,
                                   const float* __restrict__ kcL,  // [NKV][MAXSEQ][HD]
                                   const float* __restrict__ kn,   // [NKV][HD]
                                   const int* __restrict__ clen,
                                   float* __restrict__ scores) {   // [NH][MAXSEQ]
    const int lane = threadIdx.x & 31;
    const int wave = threadIdx.x >> 5;
    const int g    = blockIdx.y;
    const int tile = blockIdx.x * 8 + wave;
    const int t0   = tile * 16;
    const int half = lane >> 4;
    const int m    = lane & 15;   // A row for this lane
    const int n    = lane & 15;   // B column (timestep within tile)
    const int cl   = clen[0];

    v16h A0 = {}; v16h A1 = {};
    if (m < 4) {
        const float* qr = q + (g * 4 + m) * HD;
        #pragma unroll
        for (int e = 0; e < 16; ++e) {
            int k = (e < 8) ? (half * 8 + e) : (16 + half * 8 + (e - 8));
            A0[e] = (_Float16)qr[k];
            A1[e] = (_Float16)qr[32 + k];
        }
    }
    int t = t0 + n;
    const float* src = (t == cl) ? (kn + g * HD)
                                 : (kcL + ((size_t)g * MAXSEQ + t) * HD);
    v16h B0, B1;
    #pragma unroll
    for (int e = 0; e < 16; ++e) {
        int kb = (e < 8) ? (half * 8 + e) : (16 + half * 8 + (e - 8));
        B0[e] = (_Float16)src[kb];
        B1[e] = (_Float16)src[32 + kb];
    }
    // EXEC all-1s at these call sites (no divergence wraps the WMMA itself).
    v8f c = {};
    c = __builtin_amdgcn_wmma_f32_16x16x32_f16(false, A0, false, B0, (short)0, c, false, false);
    c = __builtin_amdgcn_wmma_f32_16x16x32_f16(false, A1, false, B1, (short)0, c, false, false);

    if (half == 0) {
        #pragma unroll
        for (int v = 0; v < 4; ++v)
            scores[(size_t)(g * 4 + v) * MAXSEQ + t] = c[v] * 0.125f; // 1/sqrt(64)
    }
}

__global__ void softmax_kernel(float* __restrict__ sc, const int* __restrict__ clen) {
    __shared__ float red[256];
    const int h = blockIdx.x, tid = threadIdx.x;
    const int T = clen[0] + 1;
    float* row = sc + (size_t)h * MAXSEQ;
    float mx = -1e30f;
    for (int t = tid; t < T; t += 256) mx = fmaxf(mx, row[t]);
    red[tid] = mx; __syncthreads();
    for (int s = 128; s > 0; s >>= 1) {
        if (tid < s) red[tid] = fmaxf(red[tid], red[tid + s]);
        __syncthreads();
    }
    mx = red[0]; __syncthreads();
    float sum = 0.f;
    for (int t = tid; t < T; t += 256) { float e = expf(row[t] - mx); row[t] = e; sum += e; }
    red[tid] = sum; __syncthreads();
    for (int s = 128; s > 0; s >>= 1) {
        if (tid < s) red[tid] += red[tid + s];
        __syncthreads();
    }
    float inv = 1.0f / red[0];
    for (int t = tid; t < T; t += 256) row[t] *= inv;
}

// o[h][d] = sum_t p[h][t] * V[g(h)][t][d]   (t == cache_len -> fresh v)
// 256 threads = 16 t-slices x 16 float4 d-lanes: lanes 0..15 of a wave stream one
// contiguous 64-float V row as b128 loads.
__global__ void attn_v_kernel(const float* __restrict__ probs,
                              const float* __restrict__ vcL,   // [NKV][MAXSEQ][HD]
                              const float* __restrict__ vn,    // [NKV][HD]
                              const int* __restrict__ clen,
                              float* __restrict__ o) {         // [NH*HD]
    __shared__ float4 red[256];
    const int h = blockIdx.x, tid = threadIdx.x;
    const int dq = tid & 15, sub = tid >> 4;
    const int g = h >> 2;
    const int cl = clen[0], T = cl + 1;
    const float* p = probs + (size_t)h * MAXSEQ;
    float4 acc = make_float4(0.f, 0.f, 0.f, 0.f);
    for (int t = sub; t < T; t += 16) {
        const float* vr = (t == cl) ? (vn + g * HD)
                                    : (vcL + ((size_t)g * MAXSEQ + t) * HD);
        float4 vv = *(const float4*)(vr + dq * 4);
        float pw = p[t];
        acc.x = fmaf(pw, vv.x, acc.x);
        acc.y = fmaf(pw, vv.y, acc.y);
        acc.z = fmaf(pw, vv.z, acc.z);
        acc.w = fmaf(pw, vv.w, acc.w);
    }
    red[tid] = acc; __syncthreads();
    if (sub == 0) {
        float4 s = red[dq];
        #pragma unroll
        for (int ss = 1; ss < 16; ++ss) {
            float4 r = red[dq + 16 * ss];
            s.x += r.x; s.y += r.y; s.z += r.z; s.w += r.w;
        }
        *(float4*)(o + h * HD + dq * 4) = s;
    }
}

// ---------------- host side ----------------

static void run_gemv(const float* x, const float* W, float* part, float* out,
                     int K, int N, int accumulate, hipStream_t s) {
    // 32-way split-K for occupancy; float4 outputs per thread.
    int kChunk = (K >= 4096) ? 256 : 64;
    int nCh = K / kChunk;
    int jBlocks = (N / 4 + 255) / 256;
    gemv_partial_kernel<<<dim3(jBlocks, nCh), 256, 0, s>>>(x, W, part, K, N, kChunk);
    reduce_kernel<<<jBlocks, 256, 0, s>>>(part, out, N, nCh, accumulate);
}

extern "C" void kernel_launch(void* const* d_in, const int* in_sizes, int n_in,
                              void* d_out, int out_size, void* d_ws, size_t ws_size,
                              hipStream_t stream) {
    (void)in_sizes; (void)n_in; (void)out_size; (void)ws_size;

    const int*   ids  = (const int*)d_in[0];
    const int*   pos  = (const int*)d_in[1];
    const int*   clen = (const int*)d_in[2];
    const float* emb  = (const float*)d_in[3];
    const float* wq   = (const float*)d_in[4];
    const float* wk   = (const float*)d_in[5];
    const float* wv   = (const float*)d_in[6];
    const float* wo   = (const float*)d_in[7];
    const float* ln1  = (const float*)d_in[8];
    const float* ln2  = (const float*)d_in[9];
    const float* wg   = (const float*)d_in[10];
    const float* wu   = (const float*)d_in[11];
    const float* wd   = (const float*)d_in[12];
    const float* kc   = (const float*)d_in[13];
    const float* vc   = (const float*)d_in[14];
    const float* fln  = (const float*)d_in[15];
    const float* lmh  = (const float*)d_in[16];
    float* out = (float*)d_out;
    float* ws  = (float*)d_ws;

    float* h    = ws + OFF_H;
    float* x    = ws + OFF_X;
    float* q    = ws + OFF_Q;
    float* kn   = ws + OFF_KN;
    float* vn   = ws + OFF_VN;
    float* sc   = ws + OFF_SC;
    float* o    = ws + OFF_O;
    float* act  = ws + OFF_ACT;
    float* part = ws + OFF_PART;

    embed_kernel<<<(HID + 255) / 256, 256, 0, stream>>>(ids, emb, h);

    for (int l = 0; l < L_LAYERS; ++l) {
        const float* wq_l = wq + (size_t)l * HID * (NH * HD);
        const float* wk_l = wk + (size_t)l * HID * (NKV * HD);
        const float* wv_l = wv + (size_t)l * HID * (NKV * HD);
        const float* wo_l = wo + (size_t)l * (NH * HD) * HID;
        const float* wg_l = wg + (size_t)l * HID * INTER;
        const float* wu_l = wu + (size_t)l * HID * INTER;
        const float* wd_l = wd + (size_t)l * INTER * HID;
        const float* kc_l = kc + (size_t)l * NKV * MAXSEQ * HD;
        const float* vc_l = vc + (size_t)l * NKV * MAXSEQ * HD;

        // x = rms(h, ln1)
        rmsnorm_kernel<<<1, 256, 0, stream>>>(h, ln1 + l * HID, x, HID);
        // q/k/v projections
        run_gemv(x, wq_l, part, q,  HID, NH * HD,  0, stream);
        run_gemv(x, wk_l, part, kn, HID, NKV * HD, 0, stream);
        run_gemv(x, wv_l, part, vn, HID, NKV * HD, 0, stream);
        // rotary on q + new k
        rope_kernel<<<5, 256, 0, stream>>>(pos, q, kn);
        // scores via WMMA f16->f32
        attn_scores_kernel<<<dim3(MAXSEQ / 16 / 8, NKV), 256, 0, stream>>>(q, kc_l, kn, clen, sc);
        softmax_kernel<<<NH, 256, 0, stream>>>(sc, clen);
        attn_v_kernel<<<NH, 256, 0, stream>>>(sc, vc_l, vn, clen, o);
        // h += o @ wo
        run_gemv(o, wo_l, part, h, NH * HD, HID, 1, stream);
        // x = rms(h, ln2); h += (silu(x@wg) * (x@wu)) @ wd
        rmsnorm_kernel<<<1, 256, 0, stream>>>(h, ln2 + l * HID, x, HID);
        {
            int kChunk = 64, nCh = HID / kChunk;             // K = 2048, 32 chunks
            float* pg = part;
            float* pu = part + (size_t)nCh * INTER;
            gemv_partial_kernel<<<dim3(INTER / 4 / 256, nCh), 256, 0, stream>>>(x, wg_l, pg, HID, INTER, kChunk);
            gemv_partial_kernel<<<dim3(INTER / 4 / 256, nCh), 256, 0, stream>>>(x, wu_l, pu, HID, INTER, kChunk);
            silu_mul_reduce_kernel<<<INTER / 4 / 256, 256, 0, stream>>>(pg, pu, act, INTER, nCh);
        }
        run_gemv(act, wd_l, part, h, INTER, HID, 1, stream);
    }

    // logits = rms(h, final_ln) @ lm_head
    rmsnorm_kernel<<<1, 256, 0, stream>>>(h, fln, x, HID);
    run_gemv(x, lmh, part, out, HID, VOCAB, 0, stream);
}
